// CSAHead_34033320853848
// MI455X (gfx1250) — compile-verified, matrix operand order checked by reading, not run on hardware
//
#include <hip/hip_runtime.h>
#include <hip/hip_bf16.h>

typedef __attribute__((ext_vector_type(16))) _Float16 v16h;
typedef __attribute__((ext_vector_type(8)))  _Float16 v8h;
typedef __attribute__((ext_vector_type(8)))  float    v8f;

#define TOKN 4096   // 64*64 pixels per batch image
#define NBAT 16

// ---------------------------------------------------------------------------
// Weight folding: bn(Wx) = (Wx - m)*s + b, s = g/sqrt(v+eps)
//  -> W' = s (row) * W (stored f16), bias' = b - m*s (fp32)
// bn param layout: stacked [4][C] = {gamma, beta, mean, var}
// ---------------------------------------------------------------------------
__global__ void fold_gemm_kernel(const float* __restrict__ W, const float* __restrict__ bn,
                                 _Float16* __restrict__ Wh, float* __restrict__ bias,
                                 int O, int Cdim) {
    int o = blockIdx.x;
    float g  = bn[o];
    float be = bn[O + o];
    float mu = bn[2 * O + o];
    float va = bn[3 * O + o];
    float s = g * rsqrtf(va + 1e-5f);
    if (threadIdx.x == 0) bias[o] = be - mu * s;
    for (int c = threadIdx.x; c < Cdim; c += blockDim.x)
        Wh[(size_t)o * Cdim + c] = (_Float16)(W[(size_t)o * Cdim + c] * s);
}

// qkv: heads come from a channel shuffle of x. heads[b,h,c2] = x[b, perm] with
// J = h*64+c2, perm = (J%4)*64 + J/4. Build a permuted 384x256 matrix so the
// whole qkv projection is one dense GEMM over the un-windowed image.
// Row R = h*96+o. Column p gathers: J = (p&63)*4 + (p>>6); weight iff J>>6==h.
__global__ void fold_qkv_kernel(const float* __restrict__ qkv_w, const float* __restrict__ qkv_bn,
                                _Float16* __restrict__ Wh, float* __restrict__ bias) {
    int R = blockIdx.x;            // 0..383
    int h = R / 96, o = R % 96;
    const float* bnh = qkv_bn + (size_t)h * 4 * 96;
    float g  = bnh[o];
    float be = bnh[96 + o];
    float mu = bnh[192 + o];
    float va = bnh[288 + o];
    float s = g * rsqrtf(va + 1e-5f);
    if (threadIdx.x == 0) bias[R] = be - mu * s;
    for (int p = threadIdx.x; p < 256; p += blockDim.x) {
        int J = (p & 63) * 4 + (p >> 6);
        _Float16 v = (_Float16)0.f;
        if ((J >> 6) == h)
            v = (_Float16)(qkv_w[((size_t)h * 96 + o) * 64 + (J & 63)] * s);
        Wh[(size_t)R * 256 + p] = v;
    }
}

__global__ void fold_dw_kernel(const float* __restrict__ W, const float* __restrict__ bn,
                               float* __restrict__ Wd, float* __restrict__ bias,
                               int C, int taps) {
    int c = blockIdx.x;
    float g  = bn[c];
    float be = bn[C + c];
    float mu = bn[2 * C + c];
    float va = bn[3 * C + c];
    float s = g * rsqrtf(va + 1e-5f);
    if (threadIdx.x == 0) bias[c] = be - mu * s;
    for (int t = threadIdx.x; t < taps; t += blockDim.x)
        Wd[c * taps + t] = W[c * taps + t] * s;
}

// ---------------------------------------------------------------------------
// Depthwise 3x3 conv + folded BN + residual: Y = X + conv'(X) + bias
// Halo tile (6 rows x 64 cols fp32) staged into LDS with the CDNA5 async
// copy path (GLOBAL_LOAD_ASYNC_TO_LDS_B128, tracked by ASYNCcnt), so the
// conv reads each global element exactly once. Off-image rows are zeroed
// with plain DS stores; both are fenced by s_wait_asynccnt + barrier.
// ---------------------------------------------------------------------------
__global__ __launch_bounds__(256) void dwconv3_kernel(const float* __restrict__ X,
                                                      const float* __restrict__ w9,
                                                      const float* __restrict__ bias,
                                                      float* __restrict__ Y) {
    __shared__ float tile[6][64];

    const int b   = blockIdx.z, c = blockIdx.y;
    const int py0 = blockIdx.x * 4;               // 4 output rows per block
    const int tid = threadIdx.x;
    const float* xc = X + ((size_t)b * 256 + c) * TOKN;

    // stage input rows py0-1 .. py0+4 (6 rows x 256B; 16B chunk per thread)
    if (tid < 96) {
        int r  = tid >> 4;                        // 0..5
        int ch = tid & 15;                        // 16B chunk within row
        int gy = py0 - 1 + r;
        if (gy >= 0 && gy < 64) {
            unsigned lds_off = (unsigned)(uintptr_t)(void*)&tile[r][ch * 4];
            const float* gptr = xc + gy * 64 + ch * 4;
            asm volatile("global_load_async_to_lds_b128 %0, %1, off"
                         :: "v"(lds_off), "v"(gptr)
                         : "memory");
        } else {
#pragma unroll
            for (int i = 0; i < 4; ++i) tile[r][ch * 4 + i] = 0.f;
        }
    }
    asm volatile("s_wait_asynccnt 0x0" ::: "memory");
    __syncthreads();

    const int lr = tid >> 6;                      // 0..3 local output row
    const int px = tid & 63;
    const float* wc = w9 + c * 9;
    float s = bias[c];
#pragma unroll
    for (int dy = 0; dy < 3; ++dy)
#pragma unroll
        for (int dx = -1; dx <= 1; ++dx) {
            int xx = px + dx;
            if (xx >= 0 && xx < 64)
                s += wc[dy * 3 + (dx + 1)] * tile[lr + dy][xx];
        }
    Y[((size_t)b * 256 + c) * TOKN + (py0 + lr) * 64 + px] = tile[lr + 1][px] + s;
}

// ---------------------------------------------------------------------------
// WMMA GEMM: Out[b][o][t] = act( sum_c W'[o][c] * X[b][c][t] + bias[o] (+res) )
// Block: 256 thr (8 wave32), tile 64(O) x 64(tok), each wave two 16x16 tiles
// sharing the A fragment. K stepped by 32 with v_wmma_f32_16x16x32_f16.
// X tile staged fp32/f16 -> f16 in LDS; rows padded to 40 halves (80B, 16B
// aligned) so each B fragment is two ds_load_b128.
// ---------------------------------------------------------------------------
template <typename TIN, bool RELU, bool RES, typename TOUT>
__global__ __launch_bounds__(256) void gemm_kernel(const TIN* __restrict__ X,
                                                   const _Float16* __restrict__ W,
                                                   const float* __restrict__ bias,
                                                   const float* __restrict__ res,
                                                   TOUT* __restrict__ out,
                                                   int Odim, int Cdim) {
    __shared__ _Float16 xs[64][40];

    const int b   = blockIdx.z;
    const int t0  = blockIdx.x * 64;
    const int o0  = blockIdx.y * 64;
    const int tid = threadIdx.x;
    const int lane = tid & 31;
    const int wave = tid >> 5;
    const int rowTile = wave & 3;          // 4 row tiles
    const int colBase = (wave >> 2) * 2;   // waves 0-3 -> cols {0,1}, 4-7 -> {2,3}
    const int m    = lane & 15;
    const int half = lane >> 4;

    const TIN* Xb = X + (size_t)b * Cdim * TOKN;
    const _Float16* Wrow = W + (size_t)(o0 + rowTile * 16 + m) * Cdim;

    v8f acc[2] = {v8f{0.f,0.f,0.f,0.f,0.f,0.f,0.f,0.f},
                  v8f{0.f,0.f,0.f,0.f,0.f,0.f,0.f,0.f}};

    // staging indices (per thread): token tloc = tid>>2, k chunk = (tid&3)*8
    const int tloc = tid >> 2;
    const int kk   = (tid & 3) * 8;

    for (int k0 = 0; k0 < Cdim; k0 += 32) {
        {
            const TIN* xp = Xb + (size_t)(k0 + kk) * TOKN + (t0 + tloc);
#pragma unroll
            for (int j = 0; j < 8; ++j)
                xs[tloc][kk + j] = (_Float16)(float)xp[(size_t)j * TOKN];
        }
        __syncthreads();

        // A fragment: lane holds row m, K = {8*half+0..7} U {16+8*half+0..7}
        v8h a_lo = *(const v8h*)(Wrow + k0 + 8 * half);
        v8h a_hi = *(const v8h*)(Wrow + k0 + 16 + 8 * half);
        v16h a;
#pragma unroll
        for (int j = 0; j < 8; ++j) { a[j] = a_lo[j]; a[j + 8] = a_hi[j]; }

#pragma unroll
        for (int ct = 0; ct < 2; ++ct) {
            int tl = (colBase + ct) * 16 + m;   // token column within block
            v8h b_lo = *(const v8h*)(&xs[tl][8 * half]);
            v8h b_hi = *(const v8h*)(&xs[tl][16 + 8 * half]);
            v16h bb;
#pragma unroll
            for (int j = 0; j < 8; ++j) { bb[j] = b_lo[j]; bb[j + 8] = b_hi[j]; }
            acc[ct] = __builtin_amdgcn_wmma_f32_16x16x32_f16(
                false, a, false, bb, (short)0, acc[ct], false, false);
        }
        __syncthreads();
    }

    // Epilogue: D element (M,N): lane = 16*(M>=8) + N, vgpr r = M&7
    const float* resb = res + (size_t)b * Odim * TOKN;
    TOUT* outb = out + (size_t)b * Odim * TOKN;
#pragma unroll
    for (int ct = 0; ct < 2; ++ct) {
        int tcol = t0 + (colBase + ct) * 16 + m;
#pragma unroll
        for (int r = 0; r < 8; ++r) {
            int o = o0 + rowTile * 16 + 8 * half + r;
            float v = acc[ct][r] + bias[o];
            if constexpr (RES) v += resb[(size_t)o * TOKN + tcol];
            if constexpr (RELU) v = fmaxf(v, 0.f);
            outb[(size_t)o * TOKN + tcol] = (TOUT)v;
        }
    }
}

// ---------------------------------------------------------------------------
// Window attention core: one block per (window, batch). Loops over 4 heads.
// F = f16 qkv output [b][384][4096] on the image grid; padded window tokens
// use the per-channel constant bqkv (BN of zero input). Output: relu(attn out)
// as f16 into R[b][h*64+d][pixel] for valid pixels (consumed by proj GEMM).
// ---------------------------------------------------------------------------
__global__ __launch_bounds__(256) void attn_kernel(const _Float16* __restrict__ F,
                                                   const float* __restrict__ bqkv,
                                                   const float* __restrict__ wqdw,
                                                   const float* __restrict__ bqdw,
                                                   const float* __restrict__ attn_bias,
                                                   _Float16* __restrict__ R) {
    __shared__ float qraw[16][49];
    __shared__ float qc[16][49];
    __shared__ float kk[16][49];
    __shared__ float vv[64][49];
    __shared__ float att[49][49];
    __shared__ int   tflat[49];
    __shared__ int   tvalid[49];

    const int wj = blockIdx.x, wi = blockIdx.y, b = blockIdx.z;
    const int tid = threadIdx.x;

    if (tid < 49) {
        int py = wi * 7 + tid / 7;
        int px = wj * 7 + tid % 7;
        int v = (py < 64 && px < 64);
        tvalid[tid] = v;
        tflat[tid]  = v ? (py * 64 + px) : 0;
    }
    __syncthreads();

    for (int h = 0; h < 4; ++h) {
        const size_t base = ((size_t)b * 384 + h * 96) * TOKN;
        // stage f for this head: ch 0..15 -> q, 16..31 -> k, 32..95 -> v
        for (int idx = tid; idx < 96 * 49; idx += 256) {
            int ch = idx / 49, i = idx % 49;
            float val = tvalid[i] ? (float)F[base + (size_t)ch * TOKN + tflat[i]]
                                  : bqkv[h * 96 + ch];
            if (ch < 16)       qraw[ch][i] = val;
            else if (ch < 32)  kk[ch - 16][i] = val;
            else               vv[ch - 32][i] = val;
        }
        __syncthreads();

        // depthwise 5x5 conv on q over the 7x7 window grid (zero pad 2)
        for (int idx = tid; idx < 16 * 49; idx += 256) {
            int ch = idx / 49, i = idx % 49;
            int iy = i / 7, ix = i % 7;
            float a = bqdw[h * 16 + ch];
            const float* wc = wqdw + (size_t)(h * 16 + ch) * 25;
            for (int dy = 0; dy < 5; ++dy) {
                int yy = iy + dy - 2;
                if (yy < 0 || yy >= 7) continue;
                for (int dx = 0; dx < 5; ++dx) {
                    int xx = ix + dx - 2;
                    if (xx < 0 || xx >= 7) continue;
                    a += wc[dy * 5 + dx] * qraw[ch][yy * 7 + xx];
                }
            }
            qc[ch][i] = a;
        }
        __syncthreads();

        // scores: att[i][j] = 0.25 * <q_i, k_j> + bias[h][|dy|*7+|dx|]
        for (int idx = tid; idx < 49 * 49; idx += 256) {
            int i = idx / 49, j = idx % 49;
            float s = 0.f;
#pragma unroll
            for (int ch = 0; ch < 16; ++ch) s += qc[ch][i] * kk[ch][j];
            int iy = i / 7, ix = i % 7, jy = j / 7, jx = j % 7;
            int ady = iy > jy ? iy - jy : jy - iy;
            int adx = ix > jx ? ix - jx : jx - ix;
            att[i][j] = s * 0.25f + attn_bias[h * 49 + ady * 7 + adx];
        }
        __syncthreads();

        // softmax over keys
        if (tid < 49) {
            float mx = -1e30f;
            for (int j = 0; j < 49; ++j) mx = fmaxf(mx, att[tid][j]);
            float sum = 0.f;
            for (int j = 0; j < 49; ++j) {
                float e = __expf(att[tid][j] - mx);
                att[tid][j] = e;
                sum += e;
            }
            float inv = 1.f / sum;
            for (int j = 0; j < 49; ++j) att[tid][j] *= inv;
        }
        __syncthreads();

        // out[d][i] = sum_j v[d][j]*att[i][j]; store relu as f16 (valid only)
        for (int idx = tid; idx < 64 * 49; idx += 256) {
            int d = idx / 49, i = idx % 49;
            if (!tvalid[i]) continue;
            float s = 0.f;
            for (int j = 0; j < 49; ++j) s += vv[d][j] * att[i][j];
            s = fmaxf(s, 0.f);
            R[((size_t)b * 256 + h * 64 + d) * TOKN + tflat[i]] = (_Float16)s;
        }
        __syncthreads();
    }
}

// ---------------------------------------------------------------------------
extern "C" void kernel_launch(void* const* d_in, const int* in_sizes, int n_in,
                              void* d_out, int out_size, void* d_ws, size_t ws_size,
                              hipStream_t stream) {
    (void)in_sizes; (void)n_in; (void)out_size; (void)ws_size;

    const float* x         = (const float*)d_in[0];
    const float* dw0_w     = (const float*)d_in[1];
    const float* dw0_bn    = (const float*)d_in[2];
    const float* ffn0_w1   = (const float*)d_in[3];
    const float* ffn0_bn1  = (const float*)d_in[4];
    const float* ffn0_w2   = (const float*)d_in[5];
    const float* ffn0_bn2  = (const float*)d_in[6];
    const float* qkv_w     = (const float*)d_in[7];
    const float* qkv_bn    = (const float*)d_in[8];
    const float* qdw_w     = (const float*)d_in[9];
    const float* qdw_bn    = (const float*)d_in[10];
    const float* attn_bias = (const float*)d_in[11];
    const float* proj_w    = (const float*)d_in[12];
    const float* proj_bn   = (const float*)d_in[13];
    const float* dw1_w     = (const float*)d_in[14];
    const float* dw1_bn    = (const float*)d_in[15];
    const float* ffn1_w1   = (const float*)d_in[16];
    const float* ffn1_bn1  = (const float*)d_in[17];
    const float* ffn1_w2   = (const float*)d_in[18];
    const float* ffn1_bn2  = (const float*)d_in[19];

    char* ws = (char*)d_ws;
    size_t off = 0;
    auto take = [&](size_t bytes) -> char* {
        char* p = ws + off;
        off += (bytes + 255) & ~(size_t)255;
        return p;
    };

    _Float16* Wf01 = (_Float16*)take((size_t)512 * 256 * 2);
    _Float16* Wf02 = (_Float16*)take((size_t)256 * 512 * 2);
    _Float16* Wqkv = (_Float16*)take((size_t)384 * 256 * 2);
    _Float16* Wprj = (_Float16*)take((size_t)256 * 256 * 2);
    _Float16* Wf11 = (_Float16*)take((size_t)512 * 256 * 2);
    _Float16* Wf12 = (_Float16*)take((size_t)256 * 512 * 2);
    float* bf01 = (float*)take(512 * 4);
    float* bf02 = (float*)take(256 * 4);
    float* bqkv = (float*)take(384 * 4);
    float* bprj = (float*)take(256 * 4);
    float* bf11 = (float*)take(512 * 4);
    float* bf12 = (float*)take(256 * 4);
    float* wdw0 = (float*)take(256 * 9 * 4);
    float* bdw0 = (float*)take(256 * 4);
    float* wdw1 = (float*)take(256 * 9 * 4);
    float* bdw1 = (float*)take(256 * 4);
    float* wqdw = (float*)take(64 * 25 * 4);
    float* bqdw = (float*)take(64 * 4);

    float*    bufB = (float*)take((size_t)NBAT * 256 * TOKN * 4);    // 64 MB fp32 image
    _Float16* bufH = (_Float16*)take((size_t)NBAT * 512 * TOKN * 2); // 64 MB f16 (hidden / f)
    _Float16* bufR = (_Float16*)take((size_t)NBAT * 256 * TOKN * 2); // 32 MB f16 relu(attn out)
    float*    bufA = (float*)d_out;                                  // use d_out as ping buffer

    // ---- fold weights (tiny) ----
    fold_gemm_kernel<<<512, 256, 0, stream>>>(ffn0_w1, ffn0_bn1, Wf01, bf01, 512, 256);
    fold_gemm_kernel<<<256, 256, 0, stream>>>(ffn0_w2, ffn0_bn2, Wf02, bf02, 256, 512);
    fold_qkv_kernel<<<384, 256, 0, stream>>>(qkv_w, qkv_bn, Wqkv, bqkv);
    fold_gemm_kernel<<<256, 256, 0, stream>>>(proj_w, proj_bn, Wprj, bprj, 256, 256);
    fold_gemm_kernel<<<512, 256, 0, stream>>>(ffn1_w1, ffn1_bn1, Wf11, bf11, 512, 256);
    fold_gemm_kernel<<<256, 256, 0, stream>>>(ffn1_w2, ffn1_bn2, Wf12, bf12, 256, 512);
    fold_dw_kernel<<<256, 32, 0, stream>>>(dw0_w, dw0_bn, wdw0, bdw0, 256, 9);
    fold_dw_kernel<<<256, 32, 0, stream>>>(dw1_w, dw1_bn, wdw1, bdw1, 256, 9);
    fold_dw_kernel<<<64, 32, 0, stream>>>(qdw_w, qdw_bn, wqdw, bqdw, 64, 25);

    // ---- t1 = x + bn(dwconv3(x)) -> A ----
    dwconv3_kernel<<<dim3(16, 256, NBAT), 256, 0, stream>>>(x, wdw0, bdw0, bufA);

    // ---- FFN0: H = relu(W1 t1 + b1) f16 ; t2 = t1 + W2 H + b2 -> B ----
    gemm_kernel<float, true, false, _Float16>
        <<<dim3(64, 8, NBAT), 256, 0, stream>>>(bufA, Wf01, bf01, nullptr, bufH, 512, 256);
    gemm_kernel<_Float16, false, true, float>
        <<<dim3(64, 4, NBAT), 256, 0, stream>>>(bufH, Wf02, bf02, bufA, bufB, 256, 512);

    // ---- attention: f = Wqkv t2 + bqkv (f16, H); window core -> R; proj + res -> A ----
    gemm_kernel<float, false, false, _Float16>
        <<<dim3(64, 6, NBAT), 256, 0, stream>>>(bufB, Wqkv, bqkv, nullptr, bufH, 384, 256);
    attn_kernel<<<dim3(10, 10, NBAT), 256, 0, stream>>>(bufH, bqkv, wqdw, bqdw, attn_bias, bufR);
    gemm_kernel<_Float16, false, true, float>
        <<<dim3(64, 4, NBAT), 256, 0, stream>>>(bufR, Wprj, bprj, bufB, bufA, 256, 256);

    // ---- t4 = t3 + bn(dwconv3(t3)) -> B ----
    dwconv3_kernel<<<dim3(16, 256, NBAT), 256, 0, stream>>>(bufA, wdw1, bdw1, bufB);

    // ---- FFN1: H = relu(W1 t4 + b1) ; out = t4 + W2 H + b2 -> d_out ----
    gemm_kernel<float, true, false, _Float16>
        <<<dim3(64, 8, NBAT), 256, 0, stream>>>(bufB, Wf11, bf11, nullptr, bufH, 512, 256);
    gemm_kernel<_Float16, false, true, float>
        <<<dim3(64, 4, NBAT), 256, 0, stream>>>(bufH, Wf12, bf12, bufB, bufA, 256, 512);
}
